// BahdanauAttention_4801773437654
// MI455X (gfx1250) — matrix operand order repaired and checked by reading, not compiled, not run on hardware
//
#include <hip/hip_runtime.h>
#include <hip/hip_bf16.h>

// Problem sizes (fixed by the reference)
#define BB 32
#define SS 4096
#define HH 512
#define K2 1024   // 2H

typedef __attribute__((ext_vector_type(16))) __bf16 v16bf;
typedef __attribute__((ext_vector_type(8)))  float  v8f;

// ---------------------------------------------------------------------------
// Kernel 1: convert Ua_w [H,2H] f32 -> bf16, pre-swizzled into WMMA B-fragment
// order: dest[((kb*32 + NI)*32 + lane)*16 + e], where for the 32x16 bf16 B
// tile: k = kb*32 + 16*(lane>>4) + e, n = NI*16 + (lane&15).
// ---------------------------------------------------------------------------
__global__ void k_ua_convert(const float* __restrict__ Ua,
                             unsigned short* __restrict__ UaTf) {
  int idx  = blockIdx.x * 256 + threadIdx.x;   // 0 .. 524287
  int e    = idx & 15;
  int lane = (idx >> 4) & 31;
  int t2   = idx >> 9;
  int NI   = t2 & 31;                          // n-tile 0..31 (H/16)
  int kb   = t2 >> 5;                          // k-block 0..31 (2H/32)
  int k    = kb * 32 + (lane >> 4) * 16 + e;
  int h    = NI * 16 + (lane & 15);
  __bf16 bv = (__bf16)Ua[h * K2 + k];
  UaTf[idx] = __builtin_bit_cast(unsigned short, bv);
}

// ---------------------------------------------------------------------------
// Kernel 2: Q2[b,h] = query[b,:] @ Wa_w[h,:] + Wa_b[h] + Ua_b[h]
// ---------------------------------------------------------------------------
__global__ void k_q2(const float* __restrict__ query,
                     const float* __restrict__ Wa_w,
                     const float* __restrict__ Wa_b,
                     const float* __restrict__ Ua_b,
                     float* __restrict__ Q2) {
  __shared__ float qs[HH];
  int b = blockIdx.x, t = threadIdx.x;
  qs[t]       = query[b * HH + t];
  qs[t + 256] = query[b * HH + t + 256];
  __syncthreads();
  for (int h = t; h < HH; h += 256) {
    const float4* wr = (const float4*)(Wa_w + (size_t)h * HH);
    float acc = Wa_b[h] + Ua_b[h];
#pragma unroll 4
    for (int k = 0; k < HH / 4; ++k) {
      float4 a = wr[k];
      acc += a.x * qs[4 * k] + a.y * qs[4 * k + 1] +
             a.z * qs[4 * k + 2] + a.w * qs[4 * k + 3];
    }
    Q2[b * HH + h] = acc;
  }
}

// ---------------------------------------------------------------------------
// Kernel 3: fused scores kernel.
// Grid: B * (S/64) workgroups, 256 threads (8 wave32).
// Wave w: si = w&3 (16 s-rows), nh = w>>2 (half of H columns).
// K-loop (2H in steps of 32): keys tile async-copied to LDS (f32), converted
// to bf16 A fragments, 16x v_wmma_f32_16x16x32_bf16 per wave per step.
// Epilogue: score_s = sum_h Va[h]*tanh(Q2[h] + kproj[s,h]), reduced across
// lanes (shfl_xor width 16) and waves (ds_add_f32).
// ---------------------------------------------------------------------------
__global__ void __launch_bounds__(256) k_scores(
    const float* __restrict__ keys,
    const unsigned short* __restrict__ UaTf,
    const float* __restrict__ Q2,
    const float* __restrict__ Va_w,
    const float* __restrict__ Va_b,
    float* __restrict__ scores) {
  __shared__ float Asf[64 * 36];   // keys tile, f32, padded pitch 36
  __shared__ float Q2s[HH];
  __shared__ float Vas[HH];
  __shared__ float Ssc[64];

  const int t     = threadIdx.x;
  const int b     = blockIdx.x >> 6;
  const int s0    = (blockIdx.x & 63) << 6;
  const int lane  = t & 31;
  const int wv    = t >> 5;
  const int si    = wv & 3;        // s sub-tile (16 rows)
  const int nh    = wv >> 2;       // column half (0: h<256, 1: h>=256)
  const int lhalf = lane >> 4;

  Q2s[t]       = Q2[b * HH + t];
  Q2s[t + 256] = Q2[b * HH + t + 256];
  Vas[t]       = Va_w[t];
  Vas[t + 256] = Va_w[t + 256];
  if (t < 64) Ssc[t] = 0.0f;

  const float* kbase = keys + ((size_t)b * SS + s0) * K2;
  const int arow = t >> 2;         // 0..63
  const int acol = (t & 3) * 8;    // 0,8,16,24

  v8f acc[16];
#pragma unroll
  for (int i = 0; i < 16; ++i) acc[i] = (v8f){0, 0, 0, 0, 0, 0, 0, 0};

  const v16bf* Bv = (const v16bf*)UaTf;

  for (int kb = 0; kb < 32; ++kb) {
    __syncthreads();               // previous tile fully consumed
    {
      const float* src = kbase + (size_t)arow * K2 + kb * 32 + acol;
      unsigned int lds0 =
          (unsigned int)(unsigned long long)(void*)&Asf[arow * 36 + acol];
      unsigned long long g0 = (unsigned long long)(const void*)src;
      // CDNA5 async global->LDS copy (ASYNCcnt path), 2x16B per thread.
      asm volatile("global_load_async_to_lds_b128 %0, %1, off"
                   :: "v"(lds0), "v"(g0) : "memory");
      asm volatile("global_load_async_to_lds_b128 %0, %1, off offset:16"
                   :: "v"(lds0 + 16u), "v"(g0) : "memory");
      asm volatile("s_wait_asynccnt 0" ::: "memory");
    }
    __syncthreads();

    // Build bf16 A fragment (16x32, contiguous k = 16*lhalf + e):
    // lanes 0-15: K 0..15 ; lanes 16-31: K 16..31 (matches B packing).
    v16bf afrag;
    {
      const int mrow = si * 16 + (lane & 15);
      const float4* r0 = (const float4*)&Asf[mrow * 36 + lhalf * 16];
      const float4* r1 = (const float4*)&Asf[mrow * 36 + lhalf * 16 + 8];
      float4 f0 = r0[0], f1 = r0[1], f2 = r1[0], f3 = r1[1];
      float tv[16] = {f0.x, f0.y, f0.z, f0.w, f1.x, f1.y, f1.z, f1.w,
                      f2.x, f2.y, f2.z, f2.w, f3.x, f3.y, f3.z, f3.w};
#pragma unroll
      for (int e = 0; e < 16; ++e) afrag[e] = (__bf16)tv[e];
    }

#pragma unroll
    for (int ni = 0; ni < 16; ++ni) {
      const int NI = nh * 16 + ni;
      v16bf bfr = Bv[((size_t)kb * 32 + NI) * 32 + lane];
      acc[ni] = __builtin_amdgcn_wmma_f32_16x16x32_bf16(
          false, afrag, false, bfr, (short)0, acc[ni], false, false);
    }
  }

  // Epilogue: tanh + Va dot, reduce over h.
  float partial[8];
#pragma unroll
  for (int v = 0; v < 8; ++v) partial[v] = 0.0f;
#pragma unroll
  for (int ni = 0; ni < 16; ++ni) {
    const int h  = (nh * 16 + ni) * 16 + (lane & 15);
    const float va = Vas[h];
    const float q2 = Q2s[h];
#pragma unroll
    for (int v = 0; v < 8; ++v)
      partial[v] += va * tanhf(q2 + acc[ni][v]);
  }
#pragma unroll
  for (int v = 0; v < 8; ++v) {
    float p = partial[v];
    for (int off = 1; off < 16; off <<= 1) p += __shfl_xor(p, off, 16);
    if ((lane & 15) == 0)
      atomicAdd(&Ssc[si * 16 + lhalf * 8 + v], p);   // ds_add_f32
  }
  __syncthreads();
  if (t < 64) scores[(size_t)b * SS + s0 + t] = Ssc[t] + Va_b[0];
}

// ---------------------------------------------------------------------------
// Kernel 4: softmax over S per batch row.
// ---------------------------------------------------------------------------
__global__ void k_softmax(const float* __restrict__ scores,
                          float* __restrict__ wts) {
  __shared__ float red[256];
  int b = blockIdx.x, t = threadIdx.x;
  const float* sc = scores + (size_t)b * SS;
  float m = -3.4e38f;
  for (int i = t; i < SS; i += 256) m = fmaxf(m, sc[i]);
  red[t] = m; __syncthreads();
  for (int o = 128; o > 0; o >>= 1) {
    if (t < o) red[t] = fmaxf(red[t], red[t + o]);
    __syncthreads();
  }
  m = red[0]; __syncthreads();
  float s = 0.0f;
  float* w = wts + (size_t)b * SS;
  for (int i = t; i < SS; i += 256) {
    float e = __expf(sc[i] - m);
    w[i] = e; s += e;
  }
  red[t] = s; __syncthreads();
  for (int o = 128; o > 0; o >>= 1) {
    if (t < o) red[t] += red[t + o];
    __syncthreads();
  }
  float inv = 1.0f / red[0];
  for (int i = t; i < SS; i += 256) w[i] *= inv;
}

// ---------------------------------------------------------------------------
// Kernel 5: context[b,:] = sum_s w[b,s] * keys[b,s,:]   (keys read once)
// ---------------------------------------------------------------------------
__global__ void k_context(const float* __restrict__ keys,
                          const float* __restrict__ wts,
                          float* __restrict__ ctx) {
  int b = blockIdx.x, t = threadIdx.x;   // 256 threads x float4 = 1024 cols
  const float4* kp = (const float4*)(keys + (size_t)b * SS * K2);
  const float* w = wts + (size_t)b * SS;
  float4 acc = {0, 0, 0, 0};
#pragma unroll 4
  for (int s = 0; s < SS; ++s) {
    float ws_ = w[s];
    float4 kv = kp[(size_t)s * 256 + t];
    acc.x += ws_ * kv.x; acc.y += ws_ * kv.y;
    acc.z += ws_ * kv.z; acc.w += ws_ * kv.w;
    if (s + 16 < SS)
      __builtin_prefetch(&kp[(size_t)(s + 16) * 256 + t], 0, 1);
  }
  ((float4*)ctx)[b * 256 + t] = acc;
}

// ---------------------------------------------------------------------------
// Host launcher. Inputs (setup_inputs order):
// 0 query[B,H] 1 keys[B,S,2H] 2 Wa_w[H,H] 3 Wa_b[H] 4 Ua_w[H,2H] 5 Ua_b[H]
// 6 Va_w[1,H] 7 Va_b[1].  d_out = context[B,2H] ++ attn_weights[B,S] (f32).
// ---------------------------------------------------------------------------
extern "C" void kernel_launch(void* const* d_in, const int* in_sizes, int n_in,
                              void* d_out, int out_size, void* d_ws,
                              size_t ws_size, hipStream_t stream) {
  const float* query = (const float*)d_in[0];
  const float* keys  = (const float*)d_in[1];
  const float* Wa_w  = (const float*)d_in[2];
  const float* Wa_b  = (const float*)d_in[3];
  const float* Ua_w  = (const float*)d_in[4];
  const float* Ua_b  = (const float*)d_in[5];
  const float* Va_w  = (const float*)d_in[6];
  const float* Va_b  = (const float*)d_in[7];
  float* out = (float*)d_out;

  // Workspace layout: bf16 fragment-ordered Ua (1 MiB) | Q2 (64 KiB) | scores
  unsigned short* UaTf = (unsigned short*)d_ws;
  float* Q2  = (float*)((char*)d_ws + (1 << 20));
  float* scp = (float*)((char*)d_ws + (1 << 20) + (1 << 16));

  float* ctx = out;                 // [B, 2H] = 32768 floats
  float* wts = out + BB * K2;       // [B, S]  = 131072 floats

  k_ua_convert<<<dim3((K2 * HH) / 256), dim3(256), 0, stream>>>(Ua_w, UaTf);
  k_q2<<<dim3(BB), dim3(256), 0, stream>>>(query, Wa_w, Wa_b, Ua_b, Q2);
  k_scores<<<dim3(BB * (SS / 64)), dim3(256), 0, stream>>>(keys, UaTf, Q2,
                                                           Va_w, Va_b, scp);
  k_softmax<<<dim3(BB), dim3(256), 0, stream>>>(scp, wts);
  k_context<<<dim3(BB), dim3(256), 0, stream>>>(keys, wts, ctx);
}
